// RecurrentAnatomyGAT_35897336660283
// MI455X (gfx1250) — matrix operand order repaired and checked by reading, not compiled, not run on hardware
//
#include <hip/hip_runtime.h>
#include <hip/hip_bf16.h>
#include <float.h>
#include <math.h>

#define NNODES   20000
#define EREL     60000
#define E3       180000   // 3 * EREL
#define NCLASSES 32
#define NGRAPHS  64
#define HEADS    8
#define OUTCH    48
#define FUSED    384
#define LN_EPS   1e-5f
#define SLOPE    0.2f
#define FRAG     64       // uint4 per 16x32 bf16 fragment (32 lanes * 32B)

typedef __bf16 bf16_t;
typedef bf16_t v16bf __attribute__((ext_vector_type(16)));
typedef float  v8f   __attribute__((ext_vector_type(8)));

// f32 -> bf16 bits, round-to-nearest-even
__device__ __forceinline__ unsigned bfbits(float f) {
  union { float f; unsigned u; } v; v.f = f;
  return ((v.u + 0x7FFFu + ((v.u >> 16) & 1u)) >> 16) & 0xFFFFu;
}
__device__ __forceinline__ unsigned pk2(float f0, float f1) {
  return bfbits(f0) | (bfbits(f1) << 16);
}

// ---------------------------------------------------------------------------
// Pack A[M,K] (f32, row major) into per-wave WMMA A fragments (bf16):
// fragment f = mt*KT + kt, 64 uint4 each; lane holds 16 bf16 in ISA 7.12.2 order.
// ---------------------------------------------------------------------------
__global__ void pack_a_kernel(const float* __restrict__ A, int lda, int M, int K,
                              int KT, uint4* __restrict__ out, long total) {
  long idx = (long)blockIdx.x * blockDim.x + threadIdx.x;
  if (idx >= total) return;
  int  lane = (int)(idx & 31);
  long f    = idx >> 5;
  int  kt   = (int)(f % KT);
  int  mt   = (int)(f / KT);
  int  half = lane >> 4, ml = lane & 15;
  int  mrow = mt * 16 + ml;
  int  kbase = kt * 32;
  unsigned d[8];
  if (mrow < M && kbase + 32 <= K && (lda & 3) == 0) {
    // fast path: 4x float4, no guards
    const float* p0 = A + (long)mrow * lda + kbase + 8 * half;
    float4 u0 = *(const float4*)(p0);
    float4 u1 = *(const float4*)(p0 + 4);
    float4 u2 = *(const float4*)(p0 + 16);
    float4 u3 = *(const float4*)(p0 + 20);
    d[0] = pk2(u0.x, u0.y); d[1] = pk2(u0.z, u0.w);
    d[2] = pk2(u1.x, u1.y); d[3] = pk2(u1.z, u1.w);
    d[4] = pk2(u2.x, u2.y); d[5] = pk2(u2.z, u2.w);
    d[6] = pk2(u3.x, u3.y); d[7] = pk2(u3.z, u3.w);
  } else {
#pragma unroll
    for (int p = 0; p < 8; ++p) {
      int ka = kbase + ((p < 4) ? (8 * half + 2 * p) : (16 + 8 * half + 2 * (p - 4)));
      float f0 = (mrow < M && ka     < K) ? A[(long)mrow * lda + ka]     : 0.f;
      float f1 = (mrow < M && ka + 1 < K) ? A[(long)mrow * lda + ka + 1] : 0.f;
      d[p] = pk2(f0, f1);
    }
  }
  out[f * FRAG + lane * 2]     = make_uint4(d[0], d[1], d[2], d[3]);
  out[f * FRAG + lane * 2 + 1] = make_uint4(d[4], d[5], d[6], d[7]);
}

// ---------------------------------------------------------------------------
// Pack B[K,N] (f32, row major) into WMMA B fragments (bf16):
// fragment f = kt*NTtot + nt (kt-major so the NT fragments of one k-step are
// contiguous); lane l: col = nt*16 + (l&15), K half by l>>4, pairs per dword.
// ---------------------------------------------------------------------------
__global__ void pack_b_kernel(const float* __restrict__ B, int ldb, int K, int N,
                              int NTtot, uint4* __restrict__ out, long total) {
  long idx = (long)blockIdx.x * blockDim.x + threadIdx.x;
  if (idx >= total) return;
  int  lane = (int)(idx & 31);
  long f    = idx >> 5;
  int  nt   = (int)(f % NTtot);
  int  kt   = (int)(f / NTtot);
  int  half = lane >> 4, nl = lane & 15;
  int  ncol = nt * 16 + nl;
  int  kb0  = kt * 32 + 8 * half;
  unsigned d[8];
#pragma unroll
  for (int p = 0; p < 8; ++p) {
    int k = kb0 + ((p < 4) ? 2 * p : 16 + 2 * (p - 4));
    float f0 = (ncol < N && k     < K) ? B[(long)k * ldb + ncol]       : 0.f;
    float f1 = (ncol < N && k + 1 < K) ? B[(long)(k + 1) * ldb + ncol] : 0.f;
    d[p] = pk2(f0, f1);
  }
  out[f * FRAG + lane * 2]     = make_uint4(d[0], d[1], d[2], d[3]);
  out[f * FRAG + lane * 2 + 1] = make_uint4(d[4], d[5], d[6], d[7]);
}

// ---------------------------------------------------------------------------
// GEMM on packed fragments: one wave computes a 16 x (16*NT) strip.
// Software-pipelined: fragments for step kt+1 are loaded before the WMMAs of
// step kt issue, so the loadcnt waits can be partial instead of full drains.
// ---------------------------------------------------------------------------
template <int NT>
__global__ __launch_bounds__(32) void wmma_gemm_packed_kernel(
    const uint4* __restrict__ Apk, const uint4* __restrict__ Bpk,
    const float* __restrict__ bias, float* __restrict__ C, int ldc,
    int M, int Nout, int KT, int NTtot, int act) {
  const int lane = threadIdx.x & 31;
  const int mt = blockIdx.y, ng = blockIdx.x;

  v8f acc[NT];
#pragma unroll
  for (int t = 0; t < NT; ++t) acc[t] = (v8f){0.f,0.f,0.f,0.f,0.f,0.f,0.f,0.f};

  const uint4* aP = Apk + ((long)mt * KT) * FRAG + lane * 2;
  const uint4* bP = Bpk + ((long)ng * NT) * FRAG + lane * 2;
  const long bStep = (long)NTtot * FRAG;

  // prologue: load fragments for kt = 0
  uint4 a0 = aP[0], a1 = aP[1];
  uint4 b0[NT], b1[NT];
#pragma unroll
  for (int t = 0; t < NT; ++t) { b0[t] = bP[t * FRAG]; b1[t] = bP[t * FRAG + 1]; }

  for (int kt = 0; kt < KT - 1; ++kt) {
    aP += FRAG;
    bP += bStep;
    // prefetch next k-step
    uint4 na0 = aP[0], na1 = aP[1];
    uint4 nb0[NT], nb1[NT];
#pragma unroll
    for (int t = 0; t < NT; ++t) { nb0[t] = bP[t * FRAG]; nb1[t] = bP[t * FRAG + 1]; }
    // compute current k-step
    union { uint4 q[2]; v16bf v; } ua;
    ua.q[0] = a0; ua.q[1] = a1;
#pragma unroll
    for (int t = 0; t < NT; ++t) {
      union { uint4 q[2]; v16bf v; } ub;
      ub.q[0] = b0[t]; ub.q[1] = b1[t];
      acc[t] = __builtin_amdgcn_wmma_f32_16x16x32_bf16(
          false, ua.v, false, ub.v, (short)0, acc[t], false, false);
    }
    // rotate
    a0 = na0; a1 = na1;
#pragma unroll
    for (int t = 0; t < NT; ++t) { b0[t] = nb0[t]; b1[t] = nb1[t]; }
  }
  // epilogue: last k-step
  {
    union { uint4 q[2]; v16bf v; } ua;
    ua.q[0] = a0; ua.q[1] = a1;
#pragma unroll
    for (int t = 0; t < NT; ++t) {
      union { uint4 q[2]; v16bf v; } ub;
      ub.q[0] = b0[t]; ub.q[1] = b1[t];
      acc[t] = __builtin_amdgcn_wmma_f32_16x16x32_bf16(
          false, ua.v, false, ub.v, (short)0, acc[t], false, false);
    }
  }

  const int half = lane >> 4, ml = lane & 15;
#pragma unroll
  for (int t = 0; t < NT; ++t) {
    int col = (ng * NT + t) * 16 + ml;
    if (col < Nout) {
      float bsv = bias ? bias[col] : 0.f;
#pragma unroll
      for (int v = 0; v < 8; ++v) {
        int row = mt * 16 + v + 8 * half;
        if (row < M) {
          float val = acc[t][v] + bsv;
          if (act == 1) val = fmaxf(val, 0.f);
          C[(long)row * ldc + col] = val;
        }
      }
    }
  }
}

// ---------------------------------------------------------------------------
// Utility kernels
// ---------------------------------------------------------------------------
__global__ void fill_kernel(float* p, float v, long n) {
  long i = (long)blockIdx.x * blockDim.x + threadIdx.x;
  if (i < n) p[i] = v;
}

__global__ void count_kernel(const int* __restrict__ batch, float* __restrict__ gcnt) {
  int n = blockIdx.x * blockDim.x + threadIdx.x;
  if (n < NNODES) atomicAdd(&gcnt[batch[n]], 1.f);
}

__global__ __launch_bounds__(256) void reduce_sum_sq_kernel(
    const float* __restrict__ x, long n, float* __restrict__ out) {
  __shared__ float s1[256], s2[256];
  long i = (long)blockIdx.x * blockDim.x + threadIdx.x;
  long stride = (long)gridDim.x * blockDim.x;
  float a = 0.f, b = 0.f;
  for (; i < n; i += stride) { float v = x[i]; a += v; b += v * v; }
  s1[threadIdx.x] = a; s2[threadIdx.x] = b;
  __syncthreads();
  for (int off = 128; off > 0; off >>= 1) {
    if (threadIdx.x < off) {
      s1[threadIdx.x] += s1[threadIdx.x + off];
      s2[threadIdx.x] += s2[threadIdx.x + off];
    }
    __syncthreads();
  }
  if (threadIdx.x == 0) { atomicAdd(&out[0], s1[0]); atomicAdd(&out[1], s2[0]); }
}

__global__ void enc_norm_kernel(const float* __restrict__ h,
                                const float* __restrict__ stats,
                                const float* __restrict__ lnw,
                                const float* __restrict__ lnb,
                                float* __restrict__ out, int coloff) {
  int i = blockIdx.x * blockDim.x + threadIdx.x;
  if (i >= NNODES * 128) return;
  const float inv = 1.f / ((float)NNODES * 128.f);
  float mean  = stats[0] * inv;
  float var   = fmaxf(stats[1] * inv - mean * mean, 0.f);
  float denom = sqrtf(var) + LN_EPS;
  int n = i >> 7, c = i & 127;
  float v = (h[i] - mean) / denom;
  out[(long)n * FUSED + coloff + c] = v * lnw[c] + lnb[c];
}

// ---------------------------------------------------------------------------
// Edge kernels
// ---------------------------------------------------------------------------
__device__ __forceinline__ const int* etab(int r, const int* e0, const int* e1, const int* e2) {
  return (r == 0) ? e0 : (r == 1) ? e1 : e2;
}

__global__ void edge_alpha_kernel(const int* e0, const int* e1, const int* e2,
                                  const float* __restrict__ aq,
                                  const float* __restrict__ ak,
                                  float* __restrict__ alpha,
                                  float* __restrict__ amax) {
  int idx = blockIdx.x * blockDim.x + threadIdx.x;
  if (idx >= E3 * HEADS) return;
  int e = idx >> 3, h = idx & 7;
  int r = e / EREL, el = e - r * EREL;
  const int* ei = etab(r, e0, e1, e2);
  int s = ei[el], d = ei[EREL + el];
  float v = aq[((long)r * NNODES + d) * HEADS + h] +
            ak[((long)r * NNODES + s) * HEADS + h];
  v = (v > 0.f) ? v : SLOPE * v;
  alpha[idx] = v;
  float* addr = &amax[d * HEADS + h];
  if (v >= 0.f) atomicMax((int*)addr, __float_as_int(v));
  else          atomicMin((unsigned int*)addr, __float_as_uint(v));
}

__global__ void edge_ex_kernel(const int* e0, const int* e1, const int* e2,
                               float* __restrict__ alpha,
                               const float* __restrict__ amax,
                               float* __restrict__ den) {
  int idx = blockIdx.x * blockDim.x + threadIdx.x;
  if (idx >= E3 * HEADS) return;
  int e = idx >> 3, h = idx & 7;
  int r = e / EREL, el = e - r * EREL;
  const int* ei = etab(r, e0, e1, e2);
  int d = ei[EREL + el];
  float ex = expf(alpha[idx] - amax[d * HEADS + h]);
  alpha[idx] = ex;
  atomicAdd(&den[d * HEADS + h], ex);
}

__global__ __launch_bounds__(256) void edge_msg_kernel(
    const int* e0, const int* e1, const int* e2,
    const float* __restrict__ ex, const float* __restrict__ den,
    const float* __restrict__ xr, float* __restrict__ agg) {
  int w = (blockIdx.x * blockDim.x + threadIdx.x) >> 5;
  int lane = threadIdx.x & 31;
  if (w >= E3) return;
  int r = w / EREL, el = w - r * EREL;
  const int* ei = etab(r, e0, e1, e2);
  int s = ei[el], d = ei[EREL + el];
  const float* xrow = &xr[((long)r * NNODES + s) * FUSED];
  float a[HEADS];
#pragma unroll
  for (int h = 0; h < HEADS; ++h)
    a[h] = ex[(long)w * HEADS + h] / (den[d * HEADS + h] + 1e-16f);
  float* aggrow = &agg[(long)d * FUSED];
#pragma unroll
  for (int it = 0; it < FUSED / 32; ++it) {
    int c = it * 32 + lane;
    atomicAdd(&aggrow[c], a[c / OUTCH] * xrow[c]);
  }
}

__global__ __launch_bounds__(128) void node_finalize_kernel(
    float* __restrict__ agg, const float* __restrict__ bias,
    const float* __restrict__ hin, const int* __restrict__ batch,
    float* __restrict__ gsum, float* __restrict__ gsumsq) {
  __shared__ float s1[128], s2[128];
  int n = blockIdx.x;
  float a = 0.f, b = 0.f;
#pragma unroll
  for (int it = 0; it < 3; ++it) {
    int c = it * 128 + threadIdx.x;
    long i = (long)n * FUSED + c;
    float v = agg[i] + bias[c];
    v = (v > 0.f) ? v : (expf(v) - 1.f);
    v += hin[i];
    agg[i] = v;
    a += v; b += v * v;
  }
  s1[threadIdx.x] = a; s2[threadIdx.x] = b;
  __syncthreads();
  for (int off = 64; off > 0; off >>= 1) {
    if (threadIdx.x < off) {
      s1[threadIdx.x] += s1[threadIdx.x + off];
      s2[threadIdx.x] += s2[threadIdx.x + off];
    }
    __syncthreads();
  }
  if (threadIdx.x == 0) {
    int g = batch[n];
    atomicAdd(&gsum[g],   s1[0]);
    atomicAdd(&gsumsq[g], s2[0]);
  }
}

__global__ void graph_norm_kernel(const float* __restrict__ x,
                                  const int* __restrict__ batch,
                                  const float* __restrict__ gcnt,
                                  const float* __restrict__ gsum,
                                  const float* __restrict__ gsumsq,
                                  const float* __restrict__ w,
                                  const float* __restrict__ b,
                                  float* __restrict__ out) {
  long i = (long)blockIdx.x * blockDim.x + threadIdx.x;
  if (i >= (long)NNODES * FUSED) return;
  int n = (int)(i / FUSED), c = (int)(i % FUSED);
  int g = batch[n];
  float norm = fmaxf(gcnt[g], 1.f) * (float)FUSED;
  float mean = gsum[g] / norm;
  float var  = fmaxf(gsumsq[g] / norm - mean * mean, 0.f);
  float v = (x[i] - mean) * rsqrtf(var + LN_EPS);
  out[i] = v * w[c] + b[c];
}

__global__ void softmax_prior_kernel(const float* __restrict__ logits,
                                     float* __restrict__ prior) {
  int n = blockIdx.x * blockDim.x + threadIdx.x;
  if (n >= NNODES) return;
  float l[NCLASSES];
  float m = -FLT_MAX;
#pragma unroll
  for (int c = 0; c < NCLASSES; ++c) { l[c] = logits[(long)n * NCLASSES + c]; m = fmaxf(m, l[c]); }
  float s = 0.f;
#pragma unroll
  for (int c = 0; c < NCLASSES; ++c) { l[c] = expf(l[c] - m); s += l[c]; }
  float inv = 1.f / s, pm = 0.f;
#pragma unroll
  for (int c = 0; c < NCLASSES; ++c) {
    float p = l[c] * inv;
    prior[(long)n * (NCLASSES + 1) + c] = p;
    pm = fmaxf(pm, p);
  }
  prior[(long)n * (NCLASSES + 1) + NCLASSES] = pm;
}

// ---------------------------------------------------------------------------
// Host orchestration
// ---------------------------------------------------------------------------
extern "C" void kernel_launch(void* const* d_in, const int* in_sizes, int n_in,
                              void* d_out, int out_size, void* d_ws, size_t ws_size,
                              hipStream_t stream) {
  (void)in_sizes; (void)n_in; (void)out_size; (void)ws_size;
  const float* x_visual = (const float*)d_in[0];
  const float* x_geom   = (const float*)d_in[1];
  const float* x_prior  = (const float*)d_in[2];
  const int*   e0       = (const int*)d_in[3];
  const int*   e1       = (const int*)d_in[4];
  const int*   e2       = (const int*)d_in[5];
  const int*   batch    = (const int*)d_in[6];
  const float* vis_W = (const float*)d_in[7];  const float* vis_b = (const float*)d_in[8];
  const float* vis_lnw = (const float*)d_in[9]; const float* vis_lnb = (const float*)d_in[10];
  const float* geo_W = (const float*)d_in[11]; const float* geo_b = (const float*)d_in[12];
  const float* geo_lnw = (const float*)d_in[13]; const float* geo_lnb = (const float*)d_in[14];
  const float* pri_W = (const float*)d_in[15]; const float* pri_b = (const float*)d_in[16];
  const float* pri_lnw = (const float*)d_in[17]; const float* pri_lnb = (const float*)d_in[18];
  const float* rgat1_W = (const float*)d_in[19]; const float* rgat1_q = (const float*)d_in[20];
  const float* rgat1_k = (const float*)d_in[21]; const float* rgat1_bias = (const float*)d_in[22];
  const float* n1_w = (const float*)d_in[23]; const float* n1_b = (const float*)d_in[24];
  const float* rgat2_W = (const float*)d_in[25]; const float* rgat2_q = (const float*)d_in[26];
  const float* rgat2_k = (const float*)d_in[27]; const float* rgat2_bias = (const float*)d_in[28];
  const float* n2_w = (const float*)d_in[29]; const float* n2_b = (const float*)d_in[30];
  const float* cls_W1 = (const float*)d_in[31]; const float* cls_b1 = (const float*)d_in[32];
  const float* cls_W2 = (const float*)d_in[33]; const float* cls_b2 = (const float*)d_in[34];
  float* logits = (float*)d_out;

  // workspace bump allocator (256B aligned slots)
  float* base = (float*)d_ws;
  long   off  = 0;
  auto alloc = [&](long nflt) -> float* {
    float* p = base + off;
    off += (nflt + 63) & ~63L;
    return p;
  };
  float* h_tmp  = alloc((long)NNODES * 128);
  float* h0     = alloc((long)NNODES * FUSED);
  float* h1     = alloc((long)NNODES * FUSED);
  float* h2     = alloc((long)NNODES * FUSED);
  float* xr     = alloc(3L * NNODES * FUSED);
  float* aq     = alloc(3L * NNODES * HEADS);
  float* ak     = alloc(3L * NNODES * HEADS);
  float* alpha  = alloc((long)E3 * HEADS);
  float* amax   = alloc((long)NNODES * HEADS);
  float* den    = alloc((long)NNODES * HEADS);
  float* agg    = alloc((long)NNODES * FUSED);
  float* gsum   = alloc(NGRAPHS);
  float* gsumsq = alloc(NGRAPHS);
  float* gcnt   = alloc(NGRAPHS);
  float* estats = alloc(2);
  float* prior  = alloc((long)NNODES * (NCLASSES + 1));
  // packed fragment buffers (uint4 counts * 4 floats)
  auto allocu4 = [&](long nu4) -> uint4* { return (uint4*)alloc(nu4 * 4); };
  uint4* bpk_vis = allocu4(32L * 8 * FRAG);          // K=1024,N=128
  uint4* bpk_geo = allocu4(1L * 8 * FRAG);           // K=6,N=128
  uint4* bpk_pri = allocu4(2L * 8 * FRAG);           // K=33,N=128
  uint4* bpk_r1  = allocu4(3L * 12 * 24 * FRAG);     // 3 x (K=384,N=384)
  uint4* bpk_r2  = allocu4(3L * 12 * 24 * FRAG);
  uint4* bpk_q1  = allocu4(12L * 1 * FRAG);
  uint4* bpk_k1  = allocu4(12L * 1 * FRAG);
  uint4* bpk_q2  = allocu4(12L * 1 * FRAG);
  uint4* bpk_k2  = allocu4(12L * 1 * FRAG);
  uint4* bpk_c1  = allocu4(12L * 8 * FRAG);          // K=384,N=128
  uint4* bpk_c2  = allocu4(4L * 2 * FRAG);           // K=128,N=32
  uint4* apk_vis = allocu4(1250L * 32 * FRAG);       // x_visual
  uint4* apk_s   = allocu4(1250L * 4 * FRAG);        // small-K A (geom/prior/cls hidden)
  uint4* apk_h   = allocu4(1250L * 12 * FRAG);       // K=384 activations
  uint4* apk_xr  = allocu4(3750L * 12 * FRAG);       // xr for aq/ak

  auto FILL = [&](float* p, float v, long n) {
    fill_kernel<<<(unsigned)((n + 255) / 256), 256, 0, stream>>>(p, v, n);
  };
  auto PACK_A = [&](const float* A, int lda, int M, int K, uint4* out) {
    int KT = (K + 31) / 32, MT = (M + 15) / 16;
    long threads = (long)MT * KT * 32;
    pack_a_kernel<<<(unsigned)((threads + 255) / 256), 256, 0, stream>>>(
        A, lda, M, K, KT, out, threads);
  };
  auto PACK_B = [&](const float* B, int ldb, int K, int N, uint4* out) {
    int KT = (K + 31) / 32, NTt = (N + 15) / 16;
    long threads = (long)KT * NTt * 32;
    pack_b_kernel<<<(unsigned)((threads + 255) / 256), 256, 0, stream>>>(
        B, ldb, K, N, NTt, out, threads);
  };
  auto GEMMP = [&](const uint4* Apk, const uint4* Bpk, const float* bias, float* C,
                   int ldc, int M, int K, int Nout, int act) {
    int KT = (K + 31) / 32;
    int NTtot = (Nout + 15) / 16;
    int MT = (M + 15) / 16;
    if (NTtot % 4 == 0) {
      dim3 g(NTtot / 4, MT);
      wmma_gemm_packed_kernel<4><<<g, 32, 0, stream>>>(Apk, Bpk, bias, C, ldc, M, Nout, KT, NTtot, act);
    } else if (NTtot % 2 == 0) {
      dim3 g(NTtot / 2, MT);
      wmma_gemm_packed_kernel<2><<<g, 32, 0, stream>>>(Apk, Bpk, bias, C, ldc, M, Nout, KT, NTtot, act);
    } else {
      dim3 g(NTtot, MT);
      wmma_gemm_packed_kernel<1><<<g, 32, 0, stream>>>(Apk, Bpk, bias, C, ldc, M, Nout, KT, NTtot, act);
    }
  };
  auto ENCODER = [&](const float* x, int K, uint4* apk, const uint4* bpk, const float* bvec,
                     const float* lnw, const float* lnb, int coloff) {
    PACK_A(x, K, NNODES, K, apk);
    GEMMP(apk, bpk, bvec, h_tmp, 128, NNODES, K, 128, 1);
    FILL(estats, 0.f, 2);
    reduce_sum_sq_kernel<<<512, 256, 0, stream>>>(h_tmp, (long)NNODES * 128, estats);
    int tot = NNODES * 128;
    enc_norm_kernel<<<(tot + 255) / 256, 256, 0, stream>>>(h_tmp, estats, lnw, lnb, h0, coloff);
  };
  auto RGAT = [&](const float* hin, const uint4* bpkW, const uint4* bpkq, const uint4* bpkk,
                  const float* bias, const float* nw, const float* nb, float* hout) {
    PACK_A(hin, FUSED, NNODES, FUSED, apk_h);
    for (int r = 0; r < 3; ++r)
      GEMMP(apk_h, bpkW + (long)r * 12 * 24 * FRAG, nullptr,
            xr + (long)r * NNODES * FUSED, FUSED, NNODES, FUSED, FUSED, 0);
    PACK_A(xr, FUSED, 3 * NNODES, FUSED, apk_xr);
    GEMMP(apk_xr, bpkq, nullptr, aq, HEADS, 3 * NNODES, FUSED, HEADS, 0);
    GEMMP(apk_xr, bpkk, nullptr, ak, HEADS, 3 * NNODES, FUSED, HEADS, 0);
    FILL(amax, -FLT_MAX, (long)NNODES * HEADS);
    FILL(den, 0.f, (long)NNODES * HEADS);
    FILL(agg, 0.f, (long)NNODES * FUSED);
    int eh = E3 * HEADS;
    edge_alpha_kernel<<<(eh + 255) / 256, 256, 0, stream>>>(e0, e1, e2, aq, ak, alpha, amax);
    edge_ex_kernel<<<(eh + 255) / 256, 256, 0, stream>>>(e0, e1, e2, alpha, amax, den);
    edge_msg_kernel<<<(E3 * 32 + 255) / 256, 256, 0, stream>>>(e0, e1, e2, alpha, den, xr, agg);
    FILL(gsum, 0.f, NGRAPHS);
    FILL(gsumsq, 0.f, NGRAPHS);
    node_finalize_kernel<<<NNODES, 128, 0, stream>>>(agg, bias, hin, batch, gsum, gsumsq);
    long tot = (long)NNODES * FUSED;
    graph_norm_kernel<<<(unsigned)((tot + 255) / 256), 256, 0, stream>>>(
        agg, batch, gcnt, gsum, gsumsq, nw, nb, hout);
  };

  // pack all weights once
  PACK_B(vis_W, 128, 1024, 128, bpk_vis);
  PACK_B(geo_W, 128, 6, 128, bpk_geo);
  PACK_B(pri_W, 128, NCLASSES + 1, 128, bpk_pri);
  for (int r = 0; r < 3; ++r) {
    PACK_B(rgat1_W + (long)r * FUSED * FUSED, FUSED, FUSED, FUSED, bpk_r1 + (long)r * 12 * 24 * FRAG);
    PACK_B(rgat2_W + (long)r * FUSED * FUSED, FUSED, FUSED, FUSED, bpk_r2 + (long)r * 12 * 24 * FRAG);
  }
  PACK_B(rgat1_q, HEADS, FUSED, HEADS, bpk_q1);
  PACK_B(rgat1_k, HEADS, FUSED, HEADS, bpk_k1);
  PACK_B(rgat2_q, HEADS, FUSED, HEADS, bpk_q2);
  PACK_B(rgat2_k, HEADS, FUSED, HEADS, bpk_k2);
  PACK_B(cls_W1, 128, FUSED, 128, bpk_c1);
  PACK_B(cls_W2, NCLASSES, 128, NCLASSES, bpk_c2);

  FILL(gcnt, 0.f, NGRAPHS);
  count_kernel<<<(NNODES + 255) / 256, 256, 0, stream>>>(batch, gcnt);

  // static encoders -> h0 columns [0,128) and [128,256)
  ENCODER(x_visual, 1024, apk_vis, bpk_vis, vis_b, vis_lnw, vis_lnb, 0);
  ENCODER(x_geom, 6, apk_s, bpk_geo, geo_b, geo_lnw, geo_lnb, 128);

  const float* cur_prior = x_prior;
  for (int t = 0; t < 2; ++t) {
    ENCODER(cur_prior, NCLASSES + 1, apk_s, bpk_pri, pri_b, pri_lnw, pri_lnb, 256);
    RGAT(h0, bpk_r1, bpk_q1, bpk_k1, rgat1_bias, n1_w, n1_b, h1);
    RGAT(h1, bpk_r2, bpk_q2, bpk_k2, rgat2_bias, n2_w, n2_b, h2);
    // classifier
    PACK_A(h2, FUSED, NNODES, FUSED, apk_h);
    GEMMP(apk_h, bpk_c1, cls_b1, h_tmp, 128, NNODES, FUSED, 128, 1);
    PACK_A(h_tmp, 128, NNODES, 128, apk_s);
    GEMMP(apk_s, bpk_c2, cls_b2, logits, NCLASSES, NNODES, 128, NCLASSES, 0);
    if (t == 0) {
      softmax_prior_kernel<<<(NNODES + 255) / 256, 256, 0, stream>>>(logits, prior);
      cur_prior = prior;
    }
  }
}
